// FewShotLocalizer_62156766707868
// MI455X (gfx1250) — matrix operand order repaired and checked by reading, not compiled, not run on hardware
//
#include <hip/hip_runtime.h>
#include <hip/hip_bf16.h>
#include <stdint.h>

// ---------------- model constants ----------------
constexpr int DIMC = 160;
constexpr int NHC  = 8;
constexpr int HDC  = 20;     // DIM/NH
constexpr int BB   = 256;    // batch
constexpr int HWC  = 196;    // 14*14
constexpr int SC   = 48;     // support length
constexpr int GC   = 16;     // groupnorm groups
constexpr float EPSC = 1e-5f;

typedef __bf16 v16bf __attribute__((ext_vector_type(16)));
typedef float  v8f   __attribute__((ext_vector_type(8)));
typedef unsigned int u32x4 __attribute__((ext_vector_type(4)));
typedef int      i32x8 __attribute__((ext_vector_type(8)));
typedef int      i32x4 __attribute__((ext_vector_type(4)));

// ---------------- helpers ----------------
__device__ inline float warp_sum(float v) {
#pragma unroll
  for (int o = 16; o > 0; o >>= 1) v += __shfl_xor(v, o, 32);
  return v;
}
__device__ inline uint16_t f2bf(float f) {
  uint32_t u = __float_as_uint(f);
  uint32_t r = (u + 0x7FFFu + ((u >> 16) & 1u)) >> 16;
  return (uint16_t)r;
}
__device__ inline float geluf(float x) {
  return 0.5f * x * (1.0f + erff(x * 0.70710678118654752f));
}

// ---------------- f32 -> bf16 convert ----------------
__global__ void __launch_bounds__(256)
f2bf_kernel(const float* __restrict__ src, uint16_t* __restrict__ dst, int n) {
  int i = blockIdx.x * 256 + threadIdx.x;
  if (i < n) dst[i] = f2bf(src[i]);
}

// ---------------- NCHW -> token-major ----------------
__global__ void __launch_bounds__(256)
chw_to_tok_kernel(const float* __restrict__ src, float* __restrict__ dst) {
  size_t i = (size_t)blockIdx.x * 256 + threadIdx.x;
  size_t total = (size_t)BB * HWC * DIMC;
  if (i >= total) return;
  int c = (int)(i % DIMC);
  int t = (int)((i / DIMC) % HWC);
  int b = (int)(i / ((size_t)DIMC * HWC));
  dst[i] = src[((size_t)b * DIMC + c) * HWC + t];
}

// ---------------- LayerNorm (width 160) -> bf16 out ----------------
__global__ void __launch_bounds__(256)
ln160_kernel(const float* __restrict__ x, const float* __restrict__ g,
             const float* __restrict__ bb, uint16_t* __restrict__ outb, int R) {
  int wave = threadIdx.x >> 5, lane = threadIdx.x & 31;
  int row = blockIdx.x * 8 + wave;
  if (row >= R) return;
  const float* xr = x + (size_t)row * DIMC;
  float v[5];
  float s = 0.f;
#pragma unroll
  for (int i = 0; i < 5; ++i) { v[i] = xr[lane + 32 * i]; s += v[i]; }
  s = warp_sum(s);
  float mean = s * (1.0f / DIMC);
  float q = 0.f;
#pragma unroll
  for (int i = 0; i < 5; ++i) { float d = v[i] - mean; q += d * d; }
  q = warp_sum(q);
  float rstd = rsqrtf(q * (1.0f / DIMC) + EPSC);
  uint16_t* orow = outb + (size_t)row * DIMC;
#pragma unroll
  for (int i = 0; i < 5; ++i) {
    int c = lane + 32 * i;
    orow[c] = f2bf((v[i] - mean) * rstd * g[c] + bb[c]);
  }
}

// ---------------- WMMA GEMM with TDM A-tile staging ----------------
// out = act(A @ W^T + bias) + res
// A: (M,K) bf16 row-major, W: (N,K) bf16 row-major.
// One block = 8 waves; the 16xK A tile is DMA'd into LDS once by the Tensor
// Data Mover, then each wave computes one 16x16 tile (A from LDS, W from L2).
template <int KK>
__global__ void __launch_bounds__(256)
gemm_wmma_kernel(const uint16_t* __restrict__ A, const uint16_t* __restrict__ W,
                 const float* __restrict__ bias, const float* __restrict__ res,
                 float* __restrict__ outf, uint16_t* __restrict__ outb,
                 int M, int N, int act) {
  __shared__ uint16_t As[16 * KK];
  int m0 = blockIdx.x * 16;

  if (threadIdx.x < 32) {
    // Build TDM descriptor (D#): 2D tile, 16 rows x KK elems, 2B elements.
    uint64_t gaddr = (uint64_t)(uintptr_t)(A + (size_t)m0 * KK);
    uint32_t lds = (uint32_t)(uintptr_t)(void*)As;  // LDS byte offset (addr[31:0])
    u32x4 g0;
    g0[0] = 1u;                                    // count=1, user mode
    g0[1] = lds;                                   // lds_addr
    g0[2] = (uint32_t)gaddr;                       // global_addr[31:0]
    g0[3] = (uint32_t)(gaddr >> 32) | (2u << 30);  // global_addr[56:32], type=2
    i32x8 g1;
    g1[0] = 0x00010000;          // workgroup_mask=0, data_size=1 (2 bytes)
    g1[1] = (int)(KK << 16);     // tensor_dim0[15:0] = KK
    g1[2] = (int)(16 << 16);     // tensor_dim1[15:0] = 16
    g1[3] = (int)(KK << 16);     // tile_dim0 = KK
    g1[4] = 16;                  // tile_dim1 = 16
    g1[5] = KK;                  // tensor_dim0_stride[31:0] = KK (row stride)
    g1[6] = 0;
    g1[7] = 0;
    i32x4 g2 = {0, 0, 0, 0};
    i32x4 g3 = {0, 0, 0, 0};
    i32x8 g4 = {0, 0, 0, 0, 0, 0, 0, 0};
    __builtin_amdgcn_tensor_load_to_lds(g0, g1, g2, g3, g4, 0);
    __builtin_amdgcn_s_wait_tensorcnt(0);
  }
  __syncthreads();

  int wave = threadIdx.x >> 5, lane = threadIdx.x & 31;
  int ntiles = N >> 4;
  int ntile = blockIdx.y * 8 + wave;
  if (ntile >= ntiles) ntile = ntiles - 1;  // tail waves duplicate last tile (benign)
  int n0 = ntile * 16;
  int l15 = lane & 15, grp = lane >> 4;
  const uint16_t* wrow = W + (size_t)(n0 + l15) * KK;
  const uint16_t* arow = As + (size_t)l15 * KK;
  v8f acc = {};
#pragma unroll
  for (int k0 = 0; k0 < KK; k0 += 32) {
    union { v16bf v; uint32_t u[8]; u32x4 q[2]; } af, bf_;
    // A fragment from LDS (16-bit A-operand layout, wave32):
    //   lanes0-15: dwords0-3 -> K k0+0..7, dwords4-7 -> K k0+16..23 (lanes16-31: +8)
    af.q[0] = *(const u32x4*)(arow + k0 + grp * 8);
    af.q[1] = *(const u32x4*)(arow + k0 + 16 + grp * 8);
    // B fragment from global: VGPRs0-3 -> K k0+0..7, VGPRs4-7 -> K k0+16..23 (lanes16-31: +8)
#pragma unroll
    for (int v = 0; v < 8; ++v) {
      int kb = k0 + grp * 8 + (v < 4 ? 2 * v : 16 + 2 * (v - 4));
      bf_.u[v] = *(const uint32_t*)(wrow + kb);
    }
    acc = __builtin_amdgcn_wmma_f32_16x16x32_bf16(false, af.v, false, bf_.v,
                                                  (short)0, acc, false, false);
  }
  int n = n0 + l15;
  float bv = bias ? bias[n] : 0.0f;
#pragma unroll
  for (int v = 0; v < 8; ++v) {
    int m = m0 + grp * 8 + v;
    float x = acc[v] + bv;
    if (act == 1) x = geluf(x);
    size_t o = (size_t)m * N + n;
    if (res) x += res[o];
    if (outf) outf[o] = x;
    if (outb) outb[o] = f2bf(x);
  }
}

// ---------------- self attention (per b,h) ----------------
__global__ void __launch_bounds__(256)
self_attn_kernel(const float* __restrict__ qkv, uint16_t* __restrict__ outb) {
  __shared__ float Ks[HWC * HDC];
  __shared__ float Vs[HWC * HDC];
  int b = blockIdx.x, h = blockIdx.y, tid = threadIdx.x;
  for (int i = tid; i < HWC * HDC; i += 256) {
    int t = i / HDC, e = i % HDC;
    const float* base = qkv + ((size_t)(b * HWC + t)) * (3 * DIMC) + h * HDC + e;
    Ks[i] = base[DIMC];
    Vs[i] = base[2 * DIMC];
  }
  __syncthreads();
  int t = tid;
  if (t >= HWC) return;
  float qv[HDC];
  const float* qp = qkv + ((size_t)(b * HWC + t)) * (3 * DIMC) + h * HDC;
#pragma unroll
  for (int e = 0; e < HDC; ++e) qv[e] = qp[e];
  const float scale = 0.223606797749979f;  // 1/sqrt(20)
  float m = -1e30f;
  for (int j = 0; j < HWC; ++j) {
    float s = 0.f;
#pragma unroll
    for (int e = 0; e < HDC; ++e) s += qv[e] * Ks[j * HDC + e];
    m = fmaxf(m, s * scale);
  }
  float Z = 0.f;
  float accv[HDC] = {};
  for (int j = 0; j < HWC; ++j) {
    float s = 0.f;
#pragma unroll
    for (int e = 0; e < HDC; ++e) s += qv[e] * Ks[j * HDC + e];
    float p = __expf(s * scale - m);
    Z += p;
#pragma unroll
    for (int e = 0; e < HDC; ++e) accv[e] += p * Vs[j * HDC + e];
  }
  float iz = 1.0f / Z;
  uint16_t* op = outb + ((size_t)(b * HWC + t)) * DIMC + h * HDC;
#pragma unroll
  for (int e = 0; e < HDC; ++e) op[e] = f2bf(accv[e] * iz);
}

// ---------------- cross attention with learnable temperature ----------------
__global__ void __launch_bounds__(256)
cross_attn_kernel(const float* __restrict__ qp_, const float* __restrict__ kp,
                  const float* __restrict__ vp, const float* __restrict__ log_tau,
                  uint16_t* __restrict__ outb) {
  __shared__ float Ks[SC * HDC];
  __shared__ float Vs[SC * HDC];
  int b = blockIdx.x, h = blockIdx.y, tid = threadIdx.x;
  for (int i = tid; i < SC * HDC; i += 256) {
    int t = i / HDC, e = i % HDC;
    Ks[i] = kp[((size_t)(b * SC + t)) * DIMC + h * HDC + e];
    Vs[i] = vp[((size_t)(b * SC + t)) * DIMC + h * HDC + e];
  }
  __syncthreads();
  float tau = fminf(log1pf(__expf(log_tau[0])) + 0.5f, 2.0f);
  float scale = 0.223606797749979f / tau;
  int t = tid;
  if (t >= HWC) return;
  float qv[HDC];
  const float* qp2 = qp_ + ((size_t)(b * HWC + t)) * DIMC + h * HDC;
#pragma unroll
  for (int e = 0; e < HDC; ++e) qv[e] = qp2[e];
  float m = -1e30f;
  for (int j = 0; j < SC; ++j) {
    float s = 0.f;
#pragma unroll
    for (int e = 0; e < HDC; ++e) s += qv[e] * Ks[j * HDC + e];
    m = fmaxf(m, s * scale);
  }
  float Z = 0.f;
  float accv[HDC] = {};
  for (int j = 0; j < SC; ++j) {
    float s = 0.f;
#pragma unroll
    for (int e = 0; e < HDC; ++e) s += qv[e] * Ks[j * HDC + e];
    float p = __expf(s * scale - m);
    Z += p;
#pragma unroll
    for (int e = 0; e < HDC; ++e) accv[e] += p * Vs[j * HDC + e];
  }
  float iz = 1.0f / Z;
  uint16_t* op = outb + ((size_t)(b * HWC + t)) * DIMC + h * HDC;
#pragma unroll
  for (int e = 0; e < HDC; ++e) op[e] = f2bf(accv[e] * iz);
}

// ---------------- depthwise 3x3 SAME conv (token-major) -> bf16 ----------------
__global__ void __launch_bounds__(256)
dwconv_kernel(const float* __restrict__ f, const float* __restrict__ dw,
              uint16_t* __restrict__ outb) {
  size_t i = (size_t)blockIdx.x * 256 + threadIdx.x;
  size_t total = (size_t)BB * HWC * DIMC;
  if (i >= total) return;
  int c = (int)(i % DIMC);
  int t = (int)((i / DIMC) % HWC);
  int b = (int)(i / ((size_t)DIMC * HWC));
  int y = t / 14, x = t % 14;
  float acc = 0.f;
#pragma unroll
  for (int dy = -1; dy <= 1; ++dy)
#pragma unroll
    for (int dx = -1; dx <= 1; ++dx) {
      int yy = y + dy, xx = x + dx;
      if (yy >= 0 && yy < 14 && xx >= 0 && xx < 14)
        acc += f[((size_t)b * HWC + (yy * 14 + xx)) * DIMC + c] *
               dw[c * 9 + (dy + 1) * 3 + (dx + 1)];
    }
  outb[i] = f2bf(acc);
}

// ---------------- GroupNorm + GELU (per b,g) ----------------
__global__ void __launch_bounds__(256)
gn_gelu_kernel(const float* __restrict__ x, const float* __restrict__ gg,
               const float* __restrict__ gb, float* __restrict__ out) {
  __shared__ float buf[HWC * 10];
  __shared__ float red[256];
  int b = blockIdx.x, g = blockIdx.y, tid = threadIdx.x;
  const int NEL = HWC * 10;
  float s = 0.f;
  for (int i = tid; i < NEL; i += 256) {
    int p = i / 10, cc = i % 10;
    float v = x[((size_t)b * HWC + p) * DIMC + g * 10 + cc];
    buf[i] = v;
    s += v;
  }
  red[tid] = s;
  __syncthreads();
  for (int o = 128; o > 0; o >>= 1) { if (tid < o) red[tid] += red[tid + o]; __syncthreads(); }
  float mean = red[0] / (float)NEL;
  __syncthreads();
  float q = 0.f;
  for (int i = tid; i < NEL; i += 256) { float d = buf[i] - mean; q += d * d; }
  red[tid] = q;
  __syncthreads();
  for (int o = 128; o > 0; o >>= 1) { if (tid < o) red[tid] += red[tid + o]; __syncthreads(); }
  float rstd = rsqrtf(red[0] / (float)NEL + EPSC);
  __syncthreads();
  for (int i = tid; i < NEL; i += 256) {
    int p = i / 10, cc = i % 10, c = g * 10 + cc;
    float y = (buf[i] - mean) * rstd * gg[c] + gb[c];
    out[((size_t)b * HWC + p) * DIMC + c] = geluf(y);
  }
}

// ---------------- small 1x1 heads (o-major NCHW output) ----------------
__global__ void __launch_bounds__(256)
head_kernel(const float* __restrict__ f, const float* __restrict__ Wt,
            const float* __restrict__ bias, int O, float* __restrict__ out) {
  size_t total = (size_t)BB * O * HWC;
  for (size_t i = (size_t)blockIdx.x * 256 + threadIdx.x; i < total;
       i += (size_t)gridDim.x * 256) {
    int t = (int)(i % HWC);
    int o = (int)((i / HWC) % O);
    int b = (int)(i / ((size_t)HWC * O));
    const float* fr = f + ((size_t)b * HWC + t) * DIMC;
    const float* wr = Wt + (size_t)o * DIMC;
    float sAcc = bias[o];
    for (int c = 0; c < DIMC; ++c) sAcc += fr[c] * wr[c];
    out[i] = sAcc;
  }
}

// ---------------- CLS pool (one block per batch element) ----------------
__global__ void __launch_bounds__(256)
pool_kernel(const float* __restrict__ tok, const float* __restrict__ cls,
            const float* __restrict__ lnq_g, const float* __restrict__ lnq_b,
            const float* __restrict__ lnkv_g, const float* __restrict__ lnkv_b,
            const float* __restrict__ in_w, const float* __restrict__ in_b,
            const float* __restrict__ out_w, const float* __restrict__ out_b,
            const float* __restrict__ lnf_g, const float* __restrict__ lnf_b,
            const float* __restrict__ w1, const float* __restrict__ b1,
            const float* __restrict__ w2, const float* __restrict__ b2,
            float* __restrict__ summary) {
  __shared__ float kvn[SC * DIMC];   // 7680
  __shared__ float kar[SC * DIMC];   // 7680 (k, then reused for u/o/cls1/fln/h)
  __shared__ float clsln[DIMC];
  __shared__ float qvec[DIMC];
  __shared__ float sc[NHC * SC];
  __shared__ float red[256];
  int b = blockIdx.x, tid = threadIdx.x;
  int wave = tid >> 5, lane = tid & 31;

  // 1) LayerNorm support rows -> kvn
  for (int r = wave; r < SC; r += 8) {
    const float* x = tok + ((size_t)b * SC + r) * DIMC;
    float v[5];
    float s = 0.f;
#pragma unroll
    for (int i = 0; i < 5; ++i) { v[i] = x[lane + 32 * i]; s += v[i]; }
    s = warp_sum(s);
    float mean = s * (1.0f / DIMC);
    float q = 0.f;
#pragma unroll
    for (int i = 0; i < 5; ++i) { float d = v[i] - mean; q += d * d; }
    q = warp_sum(q);
    float rstd = rsqrtf(q * (1.0f / DIMC) + EPSC);
#pragma unroll
    for (int i = 0; i < 5; ++i) {
      int c = lane + 32 * i;
      kvn[r * DIMC + c] = (v[i] - mean) * rstd * lnkv_g[c] + lnkv_b[c];
    }
  }
  // 2) LN(cls) via block reduction
  float cv = (tid < DIMC) ? cls[tid] : 0.0f;
  red[tid] = cv;
  __syncthreads();
  for (int o = 128; o > 0; o >>= 1) { if (tid < o) red[tid] += red[tid + o]; __syncthreads(); }
  float mean = red[0] * (1.0f / DIMC);
  __syncthreads();
  float dd = (tid < DIMC) ? (cv - mean) : 0.0f;
  red[tid] = dd * dd;
  __syncthreads();
  for (int o = 128; o > 0; o >>= 1) { if (tid < o) red[tid] += red[tid + o]; __syncthreads(); }
  float rstd = rsqrtf(red[0] * (1.0f / DIMC) + EPSC);
  __syncthreads();
  if (tid < DIMC) clsln[tid] = (cv - mean) * rstd * lnq_g[tid] + lnq_b[tid];
  __syncthreads();
  // 3) q = clsln @ Wq^T + bq
  for (int c = tid; c < DIMC; c += 256) {
    float s = in_b[c];
    for (int d = 0; d < DIMC; ++d) s += clsln[d] * in_w[(size_t)c * DIMC + d];
    qvec[c] = s;
  }
  __syncthreads();
  // 4) k = kvn @ Wk^T + bk
  for (int i = tid; i < SC * DIMC; i += 256) {
    int r = i / DIMC, c = i % DIMC;
    float s = in_b[DIMC + c];
    const float* wr = in_w + (size_t)(DIMC + c) * DIMC;
    for (int d = 0; d < DIMC; ++d) s += kvn[r * DIMC + d] * wr[d];
    kar[i] = s;
  }
  __syncthreads();
  // 5) scores
  for (int i = tid; i < NHC * SC; i += 256) {
    int h = i / SC, j = i % SC;
    float s = 0.f;
#pragma unroll
    for (int e = 0; e < HDC; ++e) s += qvec[h * HDC + e] * kar[j * DIMC + h * HDC + e];
    sc[i] = s * 0.223606797749979f;
  }
  __syncthreads();
  // 6) softmax per head
  if (tid < NHC) {
    float m = -1e30f;
    for (int j = 0; j < SC; ++j) m = fmaxf(m, sc[tid * SC + j]);
    float Z = 0.f;
    for (int j = 0; j < SC; ++j) { float p = __expf(sc[tid * SC + j] - m); sc[tid * SC + j] = p; Z += p; }
    float iz = 1.0f / Z;
    for (int j = 0; j < SC; ++j) sc[tid * SC + j] *= iz;
  }
  __syncthreads();
  // 7) u[h][d] = sum_j a[h][j] * kvn[j][d]   (kar reused)
  float* u = kar;
  for (int i = tid; i < NHC * DIMC; i += 256) {
    int h = i / DIMC, d = i % DIMC;
    float s = 0.f;
    for (int j = 0; j < SC; ++j) s += sc[h * SC + j] * kvn[j * DIMC + d];
    u[i] = s;
  }
  __syncthreads();
  // 8) o[c] = u[h(c)] . Wv[c] + bv[c]
  float* ovec = kar + NHC * DIMC;
  for (int c = tid; c < DIMC; c += 256) {
    int h = c / HDC;
    float s = in_b[2 * DIMC + c];
    const float* wr = in_w + (size_t)(2 * DIMC + c) * DIMC;
    for (int d = 0; d < DIMC; ++d) s += u[h * DIMC + d] * wr[d];
    ovec[c] = s;
  }
  __syncthreads();
  // 9) cls1 = cls + o @ out_w^T + out_b
  float* cls1 = kar + NHC * DIMC + DIMC;
  for (int c = tid; c < DIMC; c += 256) {
    float s = out_b[c];
    const float* wr = out_w + (size_t)c * DIMC;
    for (int d = 0; d < DIMC; ++d) s += ovec[d] * wr[d];
    cls1[c] = cls[c] + s;
  }
  __syncthreads();
  // 10) fln = LN(cls1, ln_ffn)
  float c1 = (tid < DIMC) ? cls1[tid] : 0.0f;
  red[tid] = c1;
  __syncthreads();
  for (int o = 128; o > 0; o >>= 1) { if (tid < o) red[tid] += red[tid + o]; __syncthreads(); }
  float mean2 = red[0] * (1.0f / DIMC);
  __syncthreads();
  float d2 = (tid < DIMC) ? (c1 - mean2) : 0.0f;
  red[tid] = d2 * d2;
  __syncthreads();
  for (int o = 128; o > 0; o >>= 1) { if (tid < o) red[tid] += red[tid + o]; __syncthreads(); }
  float rstd2 = rsqrtf(red[0] * (1.0f / DIMC) + EPSC);
  __syncthreads();
  float* fln = kar + NHC * DIMC + 2 * DIMC;
  if (tid < DIMC) fln[tid] = (c1 - mean2) * rstd2 * lnf_g[tid] + lnf_b[tid];
  __syncthreads();
  // 11) hidden = gelu(fln @ w1^T + b1)
  float* hb = kar + NHC * DIMC + 3 * DIMC;
  for (int i = tid; i < 4 * DIMC; i += 256) {
    float s = b1[i];
    const float* wr = w1 + (size_t)i * DIMC;
    for (int c = 0; c < DIMC; ++c) s += fln[c] * wr[c];
    hb[i] = geluf(s);
  }
  __syncthreads();
  // 12) summary = cls1 + hb @ w2^T + b2
  for (int c = tid; c < DIMC; c += 256) {
    float s = b2[c];
    const float* wr = w2 + (size_t)c * (4 * DIMC);
    for (int i = 0; i < 4 * DIMC; ++i) s += hb[i] * wr[i];
    summary[(size_t)b * DIMC + c] = cls1[c] + s;
  }
}

// ================= host orchestration =================
extern "C" void kernel_launch(void* const* d_in, const int* in_sizes, int n_in,
                              void* d_out, int out_size, void* d_ws, size_t ws_size,
                              hipStream_t stream) {
  (void)in_sizes; (void)n_in; (void)out_size; (void)ws_size;
  const float* q_feat = (const float*)d_in[0];
  const float* s_tok  = (const float*)d_in[1];
  auto P = [&](int i) { return (const float*)d_in[i]; };

  const int M1 = BB * HWC;   // 50176
  const int M2 = BB * SC;    // 12288

  // workspace layout
  char* ws = (char*)d_ws;
  size_t off = 0;
  float*    q    = (float*)(ws + off);    off += (size_t)M1 * DIMC * 4;       // f32 state
  uint16_t* xbf  = (uint16_t*)(ws + off); off += (size_t)M1 * DIMC * 2;       // LN staging bf16
  uint16_t* abf  = (uint16_t*)(ws + off); off += (size_t)M1 * DIMC * 2;       // attn out bf16
  uint16_t* kvbf = (uint16_t*)(ws + off); off += (size_t)M2 * DIMC * 2;       // support LN bf16
  uint16_t* wbf  = (uint16_t*)(ws + off); off += (size_t)921600 * 2;          // bf16 weights
  off = (off + 255) & ~(size_t)255;
  float*    qkvf = (float*)(ws + off);    off += (size_t)M1 * 3 * DIMC * 4;   // big scratch

  float* qpf = qkvf;
  float* kpf = qkvf + (size_t)M1 * DIMC;
  float* vpf = qkvf + (size_t)2 * M1 * DIMC;
  uint16_t* h1bf = (uint16_t*)qkvf;   // (M1, 640) bf16, reused region
  float* t1 = qkvf;                   // tower pointwise f32 output

  auto cvt = [&](const float* src, uint16_t* dst, int n) {
    f2bf_kernel<<<(n + 255) / 256, 256, 0, stream>>>(src, dst, n);
  };
  auto gemm = [&](const uint16_t* A, const uint16_t* W, const float* bias,
                  const float* res, float* outf, uint16_t* outb,
                  int M, int N, int K, int act) {
    int ntiles = N / 16;
    dim3 grd(M / 16, (ntiles + 7) / 8);
    if (K == 160)
      gemm_wmma_kernel<160><<<grd, 256, 0, stream>>>(A, W, bias, res, outf, outb, M, N, act);
    else
      gemm_wmma_kernel<640><<<grd, 256, 0, stream>>>(A, W, bias, res, outf, outb, M, N, act);
  };

  // --- convert weights to bf16 ---
  const size_t WL = 409600;
  for (int l = 0; l < 2; ++l) {
    int L0 = 24 + 25 * l;
    uint16_t* wb = wbf + l * WL;
    cvt(P(L0 + 22), wb + 0,      3 * DIMC * DIMC);  // in_w
    cvt(P(L0 + 24), wb + 76800,  DIMC * DIMC);      // out_w
    cvt(P(L0 + 6),  wb + 102400, DIMC * DIMC);      // q_w
    cvt(P(L0 + 1),  wb + 128000, DIMC * DIMC);      // k_w
    cvt(P(L0 + 8),  wb + 153600, DIMC * DIMC);      // v_w
    cvt(P(L0 + 4),  wb + 179200, DIMC * DIMC);      // o_w
    cvt(P(L0 + 11), wb + 204800, 4 * DIMC * DIMC);  // w1
    cvt(P(L0 + 12), wb + 307200, 4 * DIMC * DIMC);  // w2
  }
  for (int t = 0; t < 4; ++t)
    cvt(P(11 + 4 * t), wbf + 819200 + t * 25600, DIMC * DIMC);  // tower pw

  // --- pool (independent) ---
  const size_t REG_SZ = (size_t)BB * 68 * HWC;
  const size_t C1_SZ  = (size_t)BB * HWC;
  float* outp = (float*)d_out;
  pool_kernel<<<BB, 256, 0, stream>>>(
      s_tok, P(74), P(84), P(83), P(82), P(81), P(86), P(85), P(88), P(87),
      P(80), P(79), P(77), P(75), P(78), P(76), outp + REG_SZ + 2 * C1_SZ);

  // --- q_feat NCHW -> token-major ---
  chw_to_tok_kernel<<<(int)(((size_t)M1 * DIMC + 255) / 256), 256, 0, stream>>>(q_feat, q);

  // --- decoder layers ---
  for (int l = 0; l < 2; ++l) {
    int L0 = 24 + 25 * l;
    const uint16_t* wb = wbf + l * WL;
    // self-attention block
    ln160_kernel<<<M1 / 8, 256, 0, stream>>>(q, P(L0 + 18), P(L0 + 17), xbf, M1);
    gemm(xbf, wb + 0, P(L0 + 21), nullptr, qkvf, nullptr, M1, 3 * DIMC, DIMC, 0);
    self_attn_kernel<<<dim3(BB, NHC), 256, 0, stream>>>(qkvf, abf);
    gemm(abf, wb + 76800, P(L0 + 23), q, q, nullptr, M1, DIMC, DIMC, 0);
    // cross-attention block
    ln160_kernel<<<M1 / 8, 256, 0, stream>>>(q, P(L0 + 20), P(L0 + 19), xbf, M1);
    ln160_kernel<<<M2 / 8, 256, 0, stream>>>(s_tok, P(L0 + 16), P(L0 + 15), kvbf, M2);
    gemm(xbf,  wb + 102400, P(L0 + 5), nullptr, qpf, nullptr, M1, DIMC, DIMC, 0);
    gemm(kvbf, wb + 128000, P(L0 + 0), nullptr, kpf, nullptr, M2, DIMC, DIMC, 0);
    gemm(kvbf, wb + 153600, P(L0 + 7), nullptr, vpf, nullptr, M2, DIMC, DIMC, 0);
    cross_attn_kernel<<<dim3(BB, NHC), 256, 0, stream>>>(qpf, kpf, vpf, P(L0 + 2), abf);
    gemm(abf, wb + 179200, P(L0 + 3), q, q, nullptr, M1, DIMC, DIMC, 0);
    // FFN block
    ln160_kernel<<<M1 / 8, 256, 0, stream>>>(q, P(L0 + 14), P(L0 + 13), xbf, M1);
    gemm(xbf,  wb + 204800, P(L0 + 9),  nullptr, nullptr, h1bf, M1, 4 * DIMC, DIMC, 1);
    gemm(h1bf, wb + 307200, P(L0 + 10), q, q, nullptr, M1, DIMC, 4 * DIMC, 0);
  }

  // --- conv tower ---
  for (int t = 0; t < 4; ++t) {
    dwconv_kernel<<<(int)(((size_t)M1 * DIMC + 255) / 256), 256, 0, stream>>>(q, P(8 + 4 * t), xbf);
    gemm(xbf, wbf + 819200 + t * 25600, nullptr, nullptr, t1, nullptr, M1, DIMC, DIMC, 0);
    gn_gelu_kernel<<<dim3(BB, GC), 256, 0, stream>>>(t1, P(10 + 4 * t), P(9 + 4 * t), q);
  }

  // --- heads ---
  head_kernel<<<(int)((REG_SZ + 255) / 256), 256, 0, stream>>>(q, P(7), P(6), 68, outp);
  head_kernel<<<(int)((C1_SZ + 255) / 256), 256, 0, stream>>>(q, P(3), P(2), 1, outp + REG_SZ);
  head_kernel<<<(int)((C1_SZ + 255) / 256), 256, 0, stream>>>(q, P(5), P(4), 1, outp + REG_SZ + C1_SZ);
}
